// ResidueGraphModel_56453050138690
// MI455X (gfx1250) — matrix operand order repaired and checked by reading, not compiled, not run on hardware
//
#include <hip/hip_runtime.h>
#include <hip/hip_fp16.h>

// ---------------------------------------------------------------------------
// Types
// ---------------------------------------------------------------------------
typedef __attribute__((ext_vector_type(16))) _Float16 v16h;
typedef __attribute__((ext_vector_type(8)))  float    v8f;
typedef __attribute__((ext_vector_type(4)))  _Float16 h4;

union FragU { uint4 q[2]; v16h v; };

// GEMM tiling: 128x128 block tile, 8 waves, each wave 32(M) x 64(N),
// K staged in 32-wide chunks, double-buffered LDS.
#define TM 128
#define TN 128
#define TK 32
#define BSTR 136   // padded LDS row stride (halfs) for B tile: 272B, 16B-aligned

#define AOFF (TM * TK)     // halfs per A buffer
#define BOFF (TK * BSTR)   // halfs per B buffer

// CDNA5 async global->LDS staging (ASYNCcnt), if the toolchain exposes it.
#if defined(__HIP_DEVICE_COMPILE__) && __has_builtin(__builtin_amdgcn_global_load_async_to_lds_b128)
#define USE_ASYNC_STAGE 1
#else
#define USE_ASYNC_STAGE 0
#endif

// ---------------------------------------------------------------------------
// Register-staging helpers (EDGE path, and FULL fallback)
//   A chunk: 128 rows x 32 halfs; thread t -> row t/2, 16-half half-row t%2
//   B chunk:  32 rows x 128 halfs; thread t -> k = t/8, 16-half group t%8
// ---------------------------------------------------------------------------
template<bool EDGE>
__device__ __forceinline__
void load_tile_regs(const _Float16* __restrict__ A, const _Float16* __restrict__ B,
                    int tileM, int tileN, int kb, int M, int K, int N,
                    int tid, uint4 r[4])
{
    const int ar  = tid >> 1;
    const int agk = kb + (tid & 1) * 16;
    const int agr = tileM + ar;
    const int bgk = kb + (tid >> 3);
    const int bgn = tileN + (tid & 7) * 16;

    if (EDGE) {
        r[0] = r[1] = r[2] = r[3] = make_uint4(0u, 0u, 0u, 0u);
        if (agr < M && (agk + 16) <= K) {         // K is a multiple of 16
            const uint4* p = (const uint4*)(A + (size_t)agr * K + agk);
            r[0] = p[0]; r[1] = p[1];
        }
        if (bgk < K && (bgn + 16) <= N) {         // N is a multiple of 16
            const uint4* p = (const uint4*)(B + (size_t)bgk * N + bgn);
            r[2] = p[0]; r[3] = p[1];
        }
    } else {
        const uint4* pa = (const uint4*)(A + (size_t)agr * K + agk);
        r[0] = pa[0]; r[1] = pa[1];
        const uint4* pb = (const uint4*)(B + (size_t)bgk * N + bgn);
        r[2] = pb[0]; r[3] = pb[1];
    }
}

__device__ __forceinline__
void store_tile_lds(_Float16* sA, _Float16* sB, int tid, const uint4 r[4])
{
    uint4* pa = (uint4*)(sA + (tid >> 1) * TK + (tid & 1) * 16);
    pa[0] = r[0]; pa[1] = r[1];
    uint4* pb = (uint4*)(sB + (tid >> 3) * BSTR + (tid & 7) * 16);
    pb[0] = r[2]; pb[1] = r[3];
}

#if USE_ASYNC_STAGE
// Builtin signature (from hipcc diagnostic): param0 is
//   '__attribute__((__vector_size__(4*sizeof(int)))) int __device__ *'  (global AS)
// param1 is the LDS-side pointer. Match those types exactly.
typedef int v4i_b128 __attribute__((__vector_size__(4 * sizeof(int))));
typedef __attribute__((address_space(1))) v4i_b128* gptr_b128;
typedef __attribute__((address_space(3))) v4i_b128* lptr_b128;

// Issue 4 async b128 DMAs (2 for A, 2 for B) straight into LDS; no VGPR data.
__device__ __forceinline__
void async_stage_full(const _Float16* __restrict__ A, const _Float16* __restrict__ B,
                      int tileM, int tileN, int kb, int K, int N,
                      int tid, _Float16* sA, _Float16* sB)
{
    const int ar  = tid >> 1;
    const int ah  = (tid & 1) * 16;
    const _Float16* ga = A + (size_t)(tileM + ar) * K + (kb + ah);
    _Float16*       la = sA + ar * TK + ah;
    __builtin_amdgcn_global_load_async_to_lds_b128((gptr_b128)ga,       (lptr_b128)la,       0, 0);
    __builtin_amdgcn_global_load_async_to_lds_b128((gptr_b128)(ga + 8), (lptr_b128)(la + 8), 0, 0);

    const int bk = tid >> 3;
    const int bn = (tid & 7) * 16;
    const _Float16* gb = B + (size_t)(kb + bk) * N + tileN + bn;
    _Float16*       lb = sB + bk * BSTR + bn;
    __builtin_amdgcn_global_load_async_to_lds_b128((gptr_b128)gb,       (lptr_b128)lb,       0, 0);
    __builtin_amdgcn_global_load_async_to_lds_b128((gptr_b128)(gb + 8), (lptr_b128)(lb + 8), 0, 0);
}

__device__ __forceinline__ void wait_async0()
{
#if __has_builtin(__builtin_amdgcn_s_wait_asynccnt)
    __builtin_amdgcn_s_wait_asynccnt(0);
#else
    asm volatile("s_wait_asynccnt 0x0" ::: "memory");
#endif
}
#endif // USE_ASYNC_STAGE

// ---------------------------------------------------------------------------
// One K-chunk of WMMAs: 2 A-frags x 4 B-frags per wave
// ---------------------------------------------------------------------------
__device__ __forceinline__
void mma_chunk(const _Float16* cA, const _Float16* cB,
               int wm, int wn, int lrow, int koff, int lane, v8f acc[2][4])
{
    FragU a[2], b[4];
    #pragma unroll
    for (int sm = 0; sm < 2; ++sm) {
        const uint4* pa = (const uint4*)(cA + (wm + sm * 16 + lrow) * TK + koff);
        a[sm].q[0] = pa[0];
        a[sm].q[1] = pa[2];   // K +16 halfs
    }
    #pragma unroll
    for (int sn = 0; sn < 4; ++sn) {
        const _Float16* pb = cB + wn + sn * 16 + lrow;
        #pragma unroll
        for (int s = 0; s < 8; ++s) {
            b[sn].v[s]     = pb[(koff + s) * BSTR];
            b[sn].v[s + 8] = pb[(16 + koff + s) * BSTR];   // K +16
        }
    }
    #pragma unroll
    for (int sm = 0; sm < 2; ++sm)
        #pragma unroll
        for (int sn = 0; sn < 4; ++sn)
            acc[sm][sn] = __builtin_amdgcn_wmma_f32_16x16x32_f16(
                false, a[sm].v, false, b[sn].v,
                (short)0, acc[sm][sn], false, false);
}

// ---------------------------------------------------------------------------
// Epilogue: C layout — VGPR v: lanes0-15 -> M=v, lanes16-31 -> M=v+8
// ---------------------------------------------------------------------------
template<bool RELU>
__device__ __forceinline__
void epilogue(const v8f acc[2][4], const float* __restrict__ bias,
              float* __restrict__ outF, _Float16* __restrict__ outH,
              int tileM, int tileN, int wm, int wn, int lrow, int lane,
              int M, int N)
{
    #pragma unroll
    for (int sm = 0; sm < 2; ++sm) {
        #pragma unroll
        for (int sn = 0; sn < 4; ++sn) {
            const int gr0 = tileM + wm + sm * 16 + ((lane >> 4) * 8);
            const int gc  = tileN + wn + sn * 16 + lrow;
            if (gc < N) {
                const float bv = bias[gc];
                #pragma unroll
                for (int v = 0; v < 8; ++v) {
                    const int gr = gr0 + v;
                    if (gr < M) {
                        float x = acc[sm][sn][v] + bv;
                        if (RELU) x = x > 0.0f ? x : 0.0f;
                        if (outF) outF[(size_t)gr * N + gc] = x;
                        if (outH) outH[(size_t)gr * N + gc] = (_Float16)x;
                    }
                }
            }
        }
    }
}

// ---------------------------------------------------------------------------
// GEMM bodies:  out = act(A[M,K] @ B[K,N] + bias)
// ---------------------------------------------------------------------------
template<bool EDGE, bool RELU>
__device__ __forceinline__
void gemm_body_regs(const _Float16* __restrict__ A, const _Float16* __restrict__ B,
                    const float* __restrict__ bias,
                    float* __restrict__ outF, _Float16* __restrict__ outH,
                    int M, int K, int N, _Float16* sA, _Float16* sB)
{
    const int tid   = threadIdx.x;
    const int lane  = tid & 31;
    const int wave  = tid >> 5;
    const int tileM = blockIdx.x * TM;
    const int tileN = blockIdx.y * TN;
    const int wm    = (wave & 3) * 32;
    const int wn    = (wave >> 2) * 64;
    const int lrow  = lane & 15;
    const int koff  = (lane >> 4) * 8;

    v8f acc[2][4];
    #pragma unroll
    for (int i = 0; i < 2; ++i)
        #pragma unroll
        for (int j = 0; j < 4; ++j)
            #pragma unroll
            for (int e = 0; e < 8; ++e) acc[i][j][e] = 0.0f;

    uint4 r[4];
    load_tile_regs<EDGE>(A, B, tileM, tileN, 0, M, K, N, tid, r);
    store_tile_lds(sA, sB, tid, r);
    __syncthreads();

    int cur = 0;
    for (int kb = 0; kb < K; kb += TK) {
        const bool hasNext = (kb + TK) < K;
        if (hasNext)
            load_tile_regs<EDGE>(A, B, tileM, tileN, kb + TK, M, K, N, tid, r);

        mma_chunk(sA + cur * AOFF, sB + cur * BOFF, wm, wn, lrow, koff, lane, acc);

        if (hasNext) {
            store_tile_lds(sA + (cur ^ 1) * AOFF, sB + (cur ^ 1) * BOFF, tid, r);
            __syncthreads();
            cur ^= 1;
        }
    }
    epilogue<RELU>(acc, bias, outF, outH, tileM, tileN, wm, wn, lrow, lane, M, N);
}

#if USE_ASYNC_STAGE
template<bool RELU>
__device__ __forceinline__
void gemm_body_full_async(const _Float16* __restrict__ A, const _Float16* __restrict__ B,
                          const float* __restrict__ bias,
                          float* __restrict__ outF, _Float16* __restrict__ outH,
                          int M, int K, int N, _Float16* sA, _Float16* sB)
{
    const int tid   = threadIdx.x;
    const int lane  = tid & 31;
    const int wave  = tid >> 5;
    const int tileM = blockIdx.x * TM;
    const int tileN = blockIdx.y * TN;
    const int wm    = (wave & 3) * 32;
    const int wn    = (wave >> 2) * 64;
    const int lrow  = lane & 15;
    const int koff  = (lane >> 4) * 8;

    v8f acc[2][4];
    #pragma unroll
    for (int i = 0; i < 2; ++i)
        #pragma unroll
        for (int j = 0; j < 4; ++j)
            #pragma unroll
            for (int e = 0; e < 8; ++e) acc[i][j][e] = 0.0f;

    async_stage_full(A, B, tileM, tileN, 0, K, N, tid, sA, sB);
    wait_async0();
    __syncthreads();

    int cur = 0;
    for (int kb = 0; kb < K; kb += TK) {
        const bool hasNext = (kb + TK) < K;
        if (hasNext)   // DMA next chunk into the other buffer while we compute
            async_stage_full(A, B, tileM, tileN, kb + TK, K, N, tid,
                             sA + (cur ^ 1) * AOFF, sB + (cur ^ 1) * BOFF);

        mma_chunk(sA + cur * AOFF, sB + cur * BOFF, wm, wn, lrow, koff, lane, acc);

        if (hasNext) {
            wait_async0();
            __syncthreads();
            cur ^= 1;
        }
    }
    epilogue<RELU>(acc, bias, outF, outH, tileM, tileN, wm, wn, lrow, lane, M, N);
}
#endif // USE_ASYNC_STAGE

template<bool RELU>
__global__ __launch_bounds__(256)
void gemm_f16_wmma(const _Float16* __restrict__ A,
                   const _Float16* __restrict__ B,
                   const float*    __restrict__ bias,
                   float*          __restrict__ outF,
                   _Float16*       __restrict__ outH,
                   int M, int K, int N)
{
    __shared__ _Float16 sA[2 * AOFF];     // 16 KB
    __shared__ _Float16 sB[2 * BOFF];     // 17.4 KB

    const bool full = ((blockIdx.x + 1) * TM <= M) &&
                      ((blockIdx.y + 1) * TN <= N) &&
                      ((K & (TK - 1)) == 0);
    if (full) {
#if USE_ASYNC_STAGE
        gemm_body_full_async<RELU>(A, B, bias, outF, outH, M, K, N, sA, sB);
#else
        gemm_body_regs<false, RELU>(A, B, bias, outF, outH, M, K, N, sA, sB);
#endif
    } else {
        gemm_body_regs<true, RELU>(A, B, bias, outF, outH, M, K, N, sA, sB);
    }
}

// ---------------------------------------------------------------------------
// Elementwise / scatter kernels
// ---------------------------------------------------------------------------
__global__ void cvt_f32_f16(const float* __restrict__ in, _Float16* __restrict__ out,
                            size_t n4)  // n4 = n/4
{
    size_t i = (size_t)blockIdx.x * blockDim.x + threadIdx.x;
    if (i < n4) {
        const float4 v = *(const float4*)(in + i * 4);
        h4 o;
        o[0] = (_Float16)v.x; o[1] = (_Float16)v.y;
        o[2] = (_Float16)v.z; o[3] = (_Float16)v.w;
        *(h4*)(out + i * 4) = o;
    }
}

__global__ void zero_f32(float* __restrict__ p, size_t n4)
{
    size_t i = (size_t)blockIdx.x * blockDim.x + threadIdx.x;
    if (i < n4) {
        float4 z; z.x = z.y = z.z = z.w = 0.0f;
        *(float4*)(p + i * 4) = z;
    }
}

// g = x + agg  ->  f16
__global__ void add_cvt(const float* __restrict__ x, const float* __restrict__ agg,
                        _Float16* __restrict__ g, size_t n4)
{
    size_t i = (size_t)blockIdx.x * blockDim.x + threadIdx.x;
    if (i < n4) {
        const float4 a = *(const float4*)(x + i * 4);
        const float4 b = *(const float4*)(agg + i * 4);
        h4 o;
        o[0] = (_Float16)(a.x + b.x); o[1] = (_Float16)(a.y + b.y);
        o[2] = (_Float16)(a.z + b.z); o[3] = (_Float16)(a.w + b.w);
        *(h4*)(g + i * 4) = o;
    }
}

// GIN sum aggregation: one block per edge, 128 threads x 4 floats (C=512)
__global__ __launch_bounds__(128)
void scatter_add(const int* __restrict__ ei, const float* __restrict__ x,
                 float* __restrict__ agg, int E, int C)
{
    const int e   = blockIdx.x;
    const int src = ei[e];
    const int dst = ei[E + e];
    const int c   = threadIdx.x * 4;
    const float4 v = *(const float4*)(x + (size_t)src * C + c);
    float* p = agg + (size_t)dst * C + c;
    __hip_atomic_fetch_add(p + 0, v.x, __ATOMIC_RELAXED, __HIP_MEMORY_SCOPE_AGENT);
    __hip_atomic_fetch_add(p + 1, v.y, __ATOMIC_RELAXED, __HIP_MEMORY_SCOPE_AGENT);
    __hip_atomic_fetch_add(p + 2, v.z, __ATOMIC_RELAXED, __HIP_MEMORY_SCOPE_AGENT);
    __hip_atomic_fetch_add(p + 3, v.w, __ATOMIC_RELAXED, __HIP_MEMORY_SCOPE_AGENT);
}

// ---------------------------------------------------------------------------
// Host orchestration
// ---------------------------------------------------------------------------
static inline dim3 gemm_grid(int M, int N) { return dim3((M + TM - 1) / TM, (N + TN - 1) / TN); }
static inline int  ew_blocks(size_t n4)    { return (int)((n4 + 255) / 256); }

extern "C" void kernel_launch(void* const* d_in, const int* in_sizes, int n_in,
                              void* d_out, int out_size, void* d_ws, size_t ws_size,
                              hipStream_t stream)
{
    (void)n_in; (void)out_size; (void)ws_size;

    const float* X   = (const float*)d_in[0];
    const int*   EI  = (const int*)  d_in[1];
    const float* W1  = (const float*)d_in[2];  const float* b1   = (const float*)d_in[3];
    const float* W2  = (const float*)d_in[4];  const float* b2   = (const float*)d_in[5];
    const float* W3  = (const float*)d_in[6];  const float* b3   = (const float*)d_in[7];
    const float* W4  = (const float*)d_in[8];  const float* b4   = (const float*)d_in[9];
    const float* Win = (const float*)d_in[10]; const float* bin_ = (const float*)d_in[11];
    const float* Wg1 = (const float*)d_in[12]; const float* bg1  = (const float*)d_in[13];
    const float* Wg2 = (const float*)d_in[14]; const float* bg2  = (const float*)d_in[15];
    const float* Wout= (const float*)d_in[16]; const float* bout = (const float*)d_in[17];

    const int N = in_sizes[0] / 512;   // 50000 residue nodes
    const int E = in_sizes[1] / 2;     // 800000 edges
    const int D = 512, H = 1024, F = 80;

    float* out_mask = (float*)d_out;                    // [N,80]
    float* out_feat = (float*)d_out + (size_t)N * F;    // [N,80]

    // ---- workspace layout (total ~264 MiB) ----
    char* ws = (char*)d_ws;
    const size_t OFF_BIG0 = 0;                  // 100 MiB region
    const size_t OFF_BIG1 = 104857600;          // 100 MiB region
    const size_t OFF_S0   = 209715200;          // 50 MiB region
    const size_t OFF_S1   = 262144000;          // 8 MiB region
    const size_t OFF_W    = 270532608;          // f16 weights (~5.3 MiB)

    _Float16* Xh  = (_Float16*)(ws + OFF_S0);   // [N,512] f16; later reused as Gh
    _Float16* H1h = (_Float16*)(ws + OFF_BIG0); // [N,1024]
    _Float16* H2h = (_Float16*)(ws + OFF_BIG1); // [N,1024]
    _Float16* H3h = (_Float16*)(ws + OFF_BIG0); // [N,512]  (H1 dead)
    _Float16* PFh = (_Float16*)(ws + OFF_S1);   // [N,80]
    float*    Xf  = (float*)   (ws + OFF_BIG0); // [N,512] f32 in_net out (H3 dead)
    float*    AGG = (float*)   (ws + OFF_BIG1); // [N,512] f32 (H2 dead)
    _Float16* Gh  = (_Float16*)(ws + OFF_S0);   // [N,512]  (Xh dead)
    _Float16* Th  = (_Float16*)(ws + OFF_BIG0); // [N,512]  (Xf dead)
    _Float16* G2h = (_Float16*)(ws + OFF_BIG1); // [N,512]  (AGG dead)

    _Float16* W1h  = (_Float16*)(ws + OFF_W);
    _Float16* W2h  = W1h  + (size_t)512 * 1024;
    _Float16* W3h  = W2h  + (size_t)1024 * 1024;
    _Float16* W4h  = W3h  + (size_t)1024 * 512;
    _Float16* Winh = W4h  + (size_t)512 * 80;
    _Float16* Wg1h = Winh + (size_t)80 * 512;
    _Float16* Wg2h = Wg1h + (size_t)512 * 512;
    _Float16* Wouth= Wg2h + (size_t)512 * 512;

    // ---- f32 -> f16 conversions (input + all weights) ----
    {
        size_t n4;
        n4 = (size_t)N * D / 4;      cvt_f32_f16<<<ew_blocks(n4), 256, 0, stream>>>(X,   Xh,   n4);
        n4 = (size_t)512 * 1024 / 4; cvt_f32_f16<<<ew_blocks(n4), 256, 0, stream>>>(W1,  W1h,  n4);
        n4 = (size_t)1024 * 1024 / 4;cvt_f32_f16<<<ew_blocks(n4), 256, 0, stream>>>(W2,  W2h,  n4);
        n4 = (size_t)1024 * 512 / 4; cvt_f32_f16<<<ew_blocks(n4), 256, 0, stream>>>(W3,  W3h,  n4);
        n4 = (size_t)512 * 80 / 4;   cvt_f32_f16<<<ew_blocks(n4), 256, 0, stream>>>(W4,  W4h,  n4);
        n4 = (size_t)80 * 512 / 4;   cvt_f32_f16<<<ew_blocks(n4), 256, 0, stream>>>(Win, Winh, n4);
        n4 = (size_t)512 * 512 / 4;  cvt_f32_f16<<<ew_blocks(n4), 256, 0, stream>>>(Wg1, Wg1h, n4);
        n4 = (size_t)512 * 512 / 4;  cvt_f32_f16<<<ew_blocks(n4), 256, 0, stream>>>(Wg2, Wg2h, n4);
        n4 = (size_t)512 * 80 / 4;   cvt_f32_f16<<<ew_blocks(n4), 256, 0, stream>>>(Wout,Wouth,n4);
    }

    // ---- aanet_proj: 512 -> 1024 -> 1024 -> 512 -> 80 ----
    gemm_f16_wmma<true ><<<gemm_grid(N, H), 256, 0, stream>>>(Xh,  W1h, b1, nullptr,  H1h, N, D, H);
    gemm_f16_wmma<true ><<<gemm_grid(N, H), 256, 0, stream>>>(H1h, W2h, b2, nullptr,  H2h, N, H, H);
    gemm_f16_wmma<true ><<<gemm_grid(N, D), 256, 0, stream>>>(H2h, W3h, b3, nullptr,  H3h, N, H, D);
    gemm_f16_wmma<false><<<gemm_grid(N, F), 256, 0, stream>>>(H3h, W4h, b4, out_feat, PFh, N, D, F);

    // ---- in_net: 80 -> 512 (f32 out for exact-precision aggregation) ----
    gemm_f16_wmma<false><<<gemm_grid(N, D), 256, 0, stream>>>(PFh, Winh, bin_, Xf, nullptr, N, F, D);

    // ---- GIN sum aggregation ----
    {
        size_t n4 = (size_t)N * D / 4;
        zero_f32<<<ew_blocks(n4), 256, 0, stream>>>(AGG, n4);
        scatter_add<<<dim3(E), 128, 0, stream>>>(EI, Xf, AGG, E, D);
        add_cvt<<<ew_blocks(n4), 256, 0, stream>>>(Xf, AGG, Gh, n4);
    }

    // ---- GIN MLP: relu(g @ Wg1 + bg1) @ Wg2 + bg2 ----
    gemm_f16_wmma<true ><<<gemm_grid(N, D), 256, 0, stream>>>(Gh, Wg1h, bg1, nullptr, Th,  N, D, D);
    gemm_f16_wmma<false><<<gemm_grid(N, D), 256, 0, stream>>>(Th, Wg2h, bg2, nullptr, G2h, N, D, D);

    // ---- out_net: 512 -> 80 ----
    gemm_f16_wmma<false><<<gemm_grid(N, F), 256, 0, stream>>>(G2h, Wouth, bout, out_mask, nullptr, N, D, F);
}
